// TransformerBlock_90022514524893
// MI455X (gfx1250) — matrix operand order repaired and checked by reading, not compiled, not run on hardware
//
#include <hip/hip_runtime.h>

// ---------------- problem constants ----------------
constexpr int Bb  = 2;
constexpr int Ss  = 1024;
constexpr int Hh  = 1024;
constexpr int NHh = 16;
constexpr int HDd = 64;
constexpr int DFf = 4096;
constexpr int Ee  = 8;
constexpr int Kk  = 2;
constexpr int Ww  = 256;
constexpr int Mtok = Bb * Ss;               // 2048 token rows
constexpr int SLOTS = Mtok * Kk + Ee * 64;  // 4608 padded expert slots

// ---------------- WMMA types & helpers ----------------
typedef __bf16 v16bf __attribute__((ext_vector_type(16)));
typedef float  v8f   __attribute__((ext_vector_type(8)));

__device__ inline __bf16 bf0() {
    unsigned short z = 0;
    return __builtin_bit_cast(__bf16, z);
}

__device__ inline __bf16 f2bf(float f) {
    union { float f; unsigned u; } c{f};
    unsigned r = c.u + 0x7FFFu + ((c.u >> 16) & 1u);   // round-to-nearest-even
    unsigned short h = (unsigned short)(r >> 16);
    return __builtin_bit_cast(__bf16, h);
}

__device__ inline __bf16 cvtA(float f)   { return f2bf(f); }
__device__ inline __bf16 cvtA(__bf16 f)  { return f; }

__device__ inline v8f zero_v8f() {
    v8f z;
#pragma unroll
    for (int i = 0; i < 8; ++i) z[i] = 0.f;
    return z;
}

__device__ inline v16bf zero_v16bf() {
    v16bf z;
#pragma unroll
    for (int i = 0; i < 16; ++i) z[i] = bf0();
    return z;
}

// A fragment: 16(M) x 32(K) bf16 tile from LDS (row-major, leading dim ld)
__device__ inline v16bf load_a_frag(const __bf16* p, int ld) {
    const int lane = threadIdx.x & 31;
    const int m  = lane & 15;
    const int kb = (lane >> 4) * 8;          // lanes 0-15: K 0-7/16-23, 16-31: K 8-15/24-31
    v16bf a;
#pragma unroll
    for (int j = 0; j < 8; ++j) {
        const int kk = ((j >> 2) << 4) + kb + ((j & 3) << 1);
        a[2 * j]     = p[m * ld + kk];
        a[2 * j + 1] = p[m * ld + kk + 1];
    }
    return a;
}

// B fragment: 32(K) x 16(N) bf16 tile from LDS (row-major, leading dim ld)
__device__ inline v16bf load_b_frag(const __bf16* p, int ld) {
    const int lane = threadIdx.x & 31;
    const int n  = lane & 15;
    const int kb = (lane >> 4) * 8;
    v16bf b;
#pragma unroll
    for (int j = 0; j < 8; ++j) {
        const int kk = ((j >> 2) << 4) + kb + ((j & 3) << 1);
        b[2 * j]     = p[kk * ld + n];
        b[2 * j + 1] = p[(kk + 1) * ld + n];
    }
    return b;
}

// B fragment with only 16 valid K rows (upper 16 zero) — used for P@V (K padded 16->32)
__device__ inline v16bf load_b_frag16(const __bf16* p, int ld) {
    const int lane = threadIdx.x & 31;
    const int n  = lane & 15;
    const int kb = (lane >> 4) * 8;
    v16bf b = zero_v16bf();
#pragma unroll
    for (int j = 0; j < 4; ++j) {
        const int kk = kb + (j << 1);
        b[2 * j]     = p[kk * ld + n];
        b[2 * j + 1] = p[(kk + 1) * ld + n];
    }
    return b;
}

__device__ inline v8f wmma_bf16(v16bf a, v16bf b, v8f c) {
    return __builtin_amdgcn_wmma_f32_16x16x32_bf16(false, a, false, b, (short)0, c, false, false);
}

// ---------------- 1. RMSNorm (fp32 in -> bf16 out) ----------------
__global__ __launch_bounds__(256) void rmsnorm_kernel(const float* __restrict__ x,
                                                      const float* __restrict__ w,
                                                      __bf16* __restrict__ out) {
    __shared__ float red[8];
    const int row = blockIdx.x;
    const float* xr = x + (size_t)row * Hh;
    float ss = 0.f;
    for (int c = threadIdx.x; c < Hh; c += 256) { float v = xr[c]; ss += v * v; }
#pragma unroll
    for (int o = 1; o < 32; o <<= 1) ss += __shfl_xor(ss, o, 32);
    if ((threadIdx.x & 31) == 0) red[threadIdx.x >> 5] = ss;
    __syncthreads();
    float tot = 0.f;
#pragma unroll
    for (int i = 0; i < 8; ++i) tot += red[i];
    const float rs = rsqrtf(tot / (float)Hh + 1e-6f);
    for (int c = threadIdx.x; c < Hh; c += 256)
        out[(size_t)row * Hh + c] = f2bf(xr[c] * w[c] * rs);
}

// ---------------- 2. Generic WMMA GEMM: C[z] = A @ B[z] (+resid) ----------------
// Block tile 64x128, K step 32, 8 waves (2x4), wave tile 32x32 (4 wmma / step)
template <typename AT, bool RESID>
__global__ __launch_bounds__(256) void gemm_kernel(const AT* __restrict__ A,
                                                   const float* __restrict__ B0,
                                                   const float* __restrict__ B1,
                                                   const float* __restrict__ B2,
                                                   float* __restrict__ C,
                                                   const float* __restrict__ resid,
                                                   int M, int N, int K) {
    __shared__ __bf16 As[64][40];
    __shared__ __bf16 Bs[32][136];
    const float* B = (blockIdx.z == 0) ? B0 : ((blockIdx.z == 1) ? B1 : B2);
    C += (size_t)blockIdx.z * (size_t)M * (size_t)N;

    const int m0 = blockIdx.y * 64;
    const int n0 = blockIdx.x * 128;
    const int tid = threadIdx.x;
    const int wave = tid >> 5, lane = tid & 31;
    const int wm = wave >> 2, wn = wave & 3;

    v8f acc[2][2];
#pragma unroll
    for (int i = 0; i < 2; ++i)
#pragma unroll
        for (int j = 0; j < 2; ++j) acc[i][j] = zero_v8f();

    for (int k0 = 0; k0 < K; k0 += 32) {
#pragma unroll
        for (int i = 0; i < 8; ++i) {                       // A: 64x32
            const int idx = tid + i * 256;
            const int r = idx >> 5, c = idx & 31;
            As[r][c] = cvtA(A[(size_t)(m0 + r) * K + k0 + c]);
        }
#pragma unroll
        for (int i = 0; i < 16; ++i) {                      // B: 32x128
            const int idx = tid + i * 256;
            const int r = idx >> 7, c = idx & 127;
            Bs[r][c] = f2bf(B[(size_t)(k0 + r) * N + n0 + c]);
        }
        if (k0 + 32 < K)                                    // stream next weight tile
            __builtin_prefetch(&B[(size_t)(k0 + 32) * N + n0 + (tid & 31) * 4], 0, 1);
        __syncthreads();

        v16bf af[2];
        af[0] = load_a_frag(&As[wm * 32][0], 40);
        af[1] = load_a_frag(&As[wm * 32 + 16][0], 40);
#pragma unroll
        for (int j = 0; j < 2; ++j) {
            v16bf bfrag = load_b_frag(&Bs[0][wn * 32 + j * 16], 136);
            acc[0][j] = wmma_bf16(af[0], bfrag, acc[0][j]);
            acc[1][j] = wmma_bf16(af[1], bfrag, acc[1][j]);
        }
        __syncthreads();
    }

    const int n = lane & 15, mh = (lane >> 4) * 8;
#pragma unroll
    for (int i = 0; i < 2; ++i)
#pragma unroll
        for (int j = 0; j < 2; ++j)
#pragma unroll
            for (int r = 0; r < 8; ++r) {
                const int row = m0 + wm * 32 + i * 16 + mh + r;
                const int col = n0 + wn * 32 + j * 16 + n;
                float v = acc[i][j][r];
                if (RESID) v += resid[(size_t)row * N + col];
                C[(size_t)row * N + col] = v;
            }
}

// ---------------- 3. RoPE (in-place on q,k) ----------------
__global__ void rope_kernel(float* __restrict__ q, float* __restrict__ k,
                            const float* __restrict__ rot) {
    const int idx = blockIdx.x * 256 + threadIdx.x;  // B*S*NH*32 threads
    const int i  = idx & 31;
    const int hh = (idx >> 5) & (NHh - 1);
    const int s  = (idx >> 9) & (Ss - 1);
    const int b  = idx >> 19;
    const float c  = rot[(size_t)s * HDd * HDd + (size_t)(2 * i) * HDd + 2 * i];
    const float sn = rot[(size_t)s * HDd * HDd + (size_t)(2 * i + 1) * HDd + 2 * i];
    const size_t base = (((size_t)b * Ss + s) * NHh + hh) * HDd + 2 * i;
    float q0 = q[base], q1 = q[base + 1];
    q[base] = c * q0 - sn * q1;  q[base + 1] = sn * q0 + c * q1;
    float k0 = k[base], k1 = k[base + 1];
    k[base] = c * k0 - sn * k1;  k[base + 1] = sn * k0 + c * k1;
}

// ---------------- 4. Flash attention, sliding window W=256 ----------------
// one wave per 16-query tile; QK^T and P@V via WMMA bf16
__global__ __launch_bounds__(256) void attn_kernel(const float* __restrict__ q,
                                                   const float* __restrict__ k,
                                                   const float* __restrict__ v,
                                                   float* __restrict__ o) {
    __shared__ __bf16 sQ[8][16][64];
    __shared__ __bf16 sKT[8][64][16];
    __shared__ __bf16 sV[8][16][64];
    __shared__ __bf16 sP[8][16][32];

    const int wave = threadIdx.x >> 5, lane = threadIdx.x & 31;
    const int tile = blockIdx.x * 8 + wave;      // 0..63
    const int h = blockIdx.y, b = blockIdx.z;
    const int i0 = tile * 16;
    const size_t rowstride = (size_t)NHh * HDd;

    const float* qbase = q + ((size_t)(b * Ss + i0) * NHh + h) * HDd;
    for (int t = lane; t < 16 * 64; t += 32) {
        const int m = t >> 6, d = t & 63;
        sQ[wave][m][d] = f2bf(qbase[(size_t)m * rowstride + d]);
    }
    for (int t = lane; t < 16 * 16; t += 32)     // zero the padded K half of P once
        sP[wave][t >> 4][16 + (t & 15)] = bf0();

    const v16bf qa0 = load_a_frag(&sQ[wave][0][0], 64);
    const v16bf qa1 = load_a_frag(&sQ[wave][0][32], 64);

    v8f oacc[4];
#pragma unroll
    for (int t = 0; t < 4; ++t) oacc[t] = zero_v8f();
    float rm[8], rl[8];
#pragma unroll
    for (int r = 0; r < 8; ++r) { rm[r] = -1e30f; rl[r] = 0.f; }

    int lo = i0 - (Ww - 1); if (lo < 0) lo = 0;
    const int jt_lo = lo >> 4;
    const int nn = lane & 15, mh = (lane >> 4) * 8;

    for (int jt = jt_lo; jt <= tile; ++jt) {
        const int j0 = jt * 16;
        const float* kb = k + ((size_t)(b * Ss + j0) * NHh + h) * HDd;
        const float* vb = v + ((size_t)(b * Ss + j0) * NHh + h) * HDd;
        for (int t = lane; t < 16 * 64; t += 32) {
            const int n2 = t >> 6, d = t & 63;
            sKT[wave][d][n2] = f2bf(kb[(size_t)n2 * rowstride + d]);
            sV[wave][n2][d]  = f2bf(vb[(size_t)n2 * rowstride + d]);
        }
        v8f s = zero_v8f();
        s = wmma_bf16(qa0, load_b_frag(&sKT[wave][0][0], 16), s);
        s = wmma_bf16(qa1, load_b_frag(&sKT[wave][32][0], 16), s);

#pragma unroll
        for (int r = 0; r < 8; ++r) {
            const int i = i0 + mh + r, j = j0 + nn;
            float sv = s[r] * 0.125f;                       // 1/sqrt(64)
            const bool ok = (j <= i) && ((i - j) < Ww);
            sv = ok ? sv : -1e9f;
            float mx = sv;
#pragma unroll
            for (int off = 1; off < 16; off <<= 1) mx = fmaxf(mx, __shfl_xor(mx, off, 32));
            const float nm = fmaxf(rm[r], mx);
            const float sc = __expf(rm[r] - nm);
            rl[r] *= sc;
#pragma unroll
            for (int t = 0; t < 4; ++t) oacc[t][r] *= sc;
            const float p = __expf(sv - nm);
            float rs = p;
#pragma unroll
            for (int off = 1; off < 16; off <<= 1) rs += __shfl_xor(rs, off, 32);
            rl[r] += rs;
            rm[r] = nm;
            sP[wave][mh + r][nn] = f2bf(p);
        }

        const v16bf pa = load_a_frag(&sP[wave][0][0], 32);  // K=32, upper 16 zero
#pragma unroll
        for (int t = 0; t < 4; ++t) {
            v16bf vf = load_b_frag16(&sV[wave][0][t * 16], 64);
            oacc[t] = wmma_bf16(pa, vf, oacc[t]);
        }
    }

    float* ob = o + ((size_t)(b * Ss + i0) * NHh + h) * HDd;
#pragma unroll
    for (int t = 0; t < 4; ++t)
#pragma unroll
        for (int r = 0; r < 8; ++r)
            ob[(size_t)(mh + r) * rowstride + t * 16 + nn] = oacc[t][r] / rl[r];
}

// ---------------- 5. MoE routing ----------------
__global__ void init_kernel(int* counts, int* cnt2, int* tokens) {
    const int idx = blockIdx.x * 256 + threadIdx.x;
    if (idx < SLOTS) tokens[idx] = -1;
    if (idx < Ee) { counts[idx] = 0; cnt2[idx] = 0; }
}

__global__ __launch_bounds__(256) void gate_kernel(const __bf16* __restrict__ n2,
                                                   const float* __restrict__ gw,
                                                   int* __restrict__ topidx,
                                                   float* __restrict__ topg,
                                                   int* __restrict__ counts) {
    const int wave = threadIdx.x >> 5, lane = threadIdx.x & 31;
    const int t = blockIdx.x * 8 + wave;
    float part[Ee];
#pragma unroll
    for (int e = 0; e < Ee; ++e) part[e] = 0.f;
    for (int h = lane; h < Hh; h += 32) {
        const float xv = (float)n2[(size_t)t * Hh + h];
        const float* g = gw + (size_t)h * Ee;
#pragma unroll
        for (int e = 0; e < Ee; ++e) part[e] += xv * g[e];
    }
#pragma unroll
    for (int e = 0; e < Ee; ++e)
#pragma unroll
        for (int off = 1; off < 32; off <<= 1) part[e] += __shfl_xor(part[e], off, 32);
    if (lane == 0) {
        int i0 = 0; float v0 = part[0];
        for (int e = 1; e < Ee; ++e) if (part[e] > v0) { v0 = part[e]; i0 = e; }
        int i1 = (i0 == 0) ? 1 : 0; float v1 = part[i1];
        for (int e = 0; e < Ee; ++e) if (e != i0 && part[e] > v1) { v1 = part[e]; i1 = e; }
        const float e1 = __expf(v1 - v0);
        topidx[t * 2] = i0; topidx[t * 2 + 1] = i1;
        topg[t * 2] = 1.f / (1.f + e1); topg[t * 2 + 1] = e1 / (1.f + e1);
        atomicAdd(&counts[i0], 1);
        atomicAdd(&counts[i1], 1);
    }
}

__global__ void scan_kernel(const int* __restrict__ counts, int* __restrict__ off) {
    if (threadIdx.x == 0 && blockIdx.x == 0) {
        int a = 0;
        for (int e = 0; e < Ee; ++e) { off[e] = a; a += (counts[e] + 63) & ~63; }
        off[Ee] = a;
    }
}

__global__ void scatter_kernel(const int* __restrict__ topidx, const float* __restrict__ topg,
                               const int* __restrict__ off, int* __restrict__ cnt2,
                               int* __restrict__ tokens, float* __restrict__ slotgate,
                               int* __restrict__ slot_of) {
    const int t = blockIdx.x * 256 + threadIdx.x;
    if (t >= Mtok) return;
    for (int kk = 0; kk < Kk; ++kk) {
        const int e = topidx[t * 2 + kk];
        const int pos = atomicAdd(&cnt2[e], 1);
        const int gs = off[e] + pos;
        tokens[gs] = t;
        slotgate[gs] = topg[t * 2 + kk];
        slot_of[t * 2 + kk] = gs;
    }
}

// ---------------- 6. MoE FFN stage 1: h1 = silu(n@w1[e]) * (n@w3[e]) ----------------
__global__ __launch_bounds__(256) void moe1_kernel(const __bf16* __restrict__ n2,
                                                   const int* __restrict__ tokens,
                                                   const int* __restrict__ off,
                                                   const float* __restrict__ w1,
                                                   const float* __restrict__ w3,
                                                   __bf16* __restrict__ h1) {
    __shared__ __bf16 As[64][40];
    __shared__ __bf16 B1s[32][136];
    __shared__ __bf16 B3s[32][136];
    const int m0 = blockIdx.y * 64;
    if (m0 >= off[Ee]) return;
    int e = 0;
    while (e < Ee - 1 && m0 >= off[e + 1]) ++e;
    const float* b1 = w1 + (size_t)e * Hh * DFf;
    const float* b3 = w3 + (size_t)e * Hh * DFf;
    const int n0 = blockIdx.x * 128;
    const int tid = threadIdx.x;
    const int wave = tid >> 5, lane = tid & 31;
    const int wm = wave >> 2, wn = wave & 3;

    v8f a1[2][2], a3[2][2];
#pragma unroll
    for (int i = 0; i < 2; ++i)
#pragma unroll
        for (int j = 0; j < 2; ++j) { a1[i][j] = zero_v8f(); a3[i][j] = zero_v8f(); }

    for (int k0 = 0; k0 < Hh; k0 += 32) {
#pragma unroll
        for (int i = 0; i < 8; ++i) {
            const int idx = tid + i * 256;
            const int r = idx >> 5, c = idx & 31;
            const int tok = tokens[m0 + r];
            As[r][c] = (tok < 0) ? bf0() : n2[(size_t)tok * Hh + k0 + c];
        }
#pragma unroll
        for (int i = 0; i < 16; ++i) {
            const int idx = tid + i * 256;
            const int r = idx >> 7, c = idx & 127;
            B1s[r][c] = f2bf(b1[(size_t)(k0 + r) * DFf + n0 + c]);
            B3s[r][c] = f2bf(b3[(size_t)(k0 + r) * DFf + n0 + c]);
        }
        if (k0 + 32 < Hh) {
            __builtin_prefetch(&b1[(size_t)(k0 + 32) * DFf + n0 + (tid & 31) * 4], 0, 1);
            __builtin_prefetch(&b3[(size_t)(k0 + 32) * DFf + n0 + (tid & 31) * 4], 0, 1);
        }
        __syncthreads();
        v16bf af[2];
        af[0] = load_a_frag(&As[wm * 32][0], 40);
        af[1] = load_a_frag(&As[wm * 32 + 16][0], 40);
#pragma unroll
        for (int j = 0; j < 2; ++j) {
            v16bf b1f = load_b_frag(&B1s[0][wn * 32 + j * 16], 136);
            v16bf b3f = load_b_frag(&B3s[0][wn * 32 + j * 16], 136);
            a1[0][j] = wmma_bf16(af[0], b1f, a1[0][j]);
            a1[1][j] = wmma_bf16(af[1], b1f, a1[1][j]);
            a3[0][j] = wmma_bf16(af[0], b3f, a3[0][j]);
            a3[1][j] = wmma_bf16(af[1], b3f, a3[1][j]);
        }
        __syncthreads();
    }

    const int n = lane & 15, mh = (lane >> 4) * 8;
#pragma unroll
    for (int i = 0; i < 2; ++i)
#pragma unroll
        for (int j = 0; j < 2; ++j)
#pragma unroll
            for (int r = 0; r < 8; ++r) {
                const float v1 = a1[i][j][r], v3 = a3[i][j][r];
                const float hv = (v1 / (1.f + __expf(-v1))) * v3;   // silu(v1)*v3
                const int row = m0 + wm * 32 + i * 16 + mh + r;
                const int col = n0 + wn * 32 + j * 16 + n;
                h1[(size_t)row * DFf + col] = f2bf(hv);
            }
}

// ---------------- 7. MoE FFN stage 2: contrib = gate * (h1 @ w2[e]) ----------------
__global__ __launch_bounds__(256) void moe2_kernel(const __bf16* __restrict__ h1,
                                                   const int* __restrict__ off,
                                                   const float* __restrict__ w2,
                                                   const float* __restrict__ slotgate,
                                                   float* __restrict__ contrib) {
    __shared__ __bf16 As[64][40];
    __shared__ __bf16 Bs[32][136];
    const int m0 = blockIdx.y * 64;
    if (m0 >= off[Ee]) return;
    int e = 0;
    while (e < Ee - 1 && m0 >= off[e + 1]) ++e;
    const float* B = w2 + (size_t)e * DFf * Hh;
    const int n0 = blockIdx.x * 128;
    const int tid = threadIdx.x;
    const int wave = tid >> 5, lane = tid & 31;
    const int wm = wave >> 2, wn = wave & 3;

    v8f acc[2][2];
#pragma unroll
    for (int i = 0; i < 2; ++i)
#pragma unroll
        for (int j = 0; j < 2; ++j) acc[i][j] = zero_v8f();

    for (int k0 = 0; k0 < DFf; k0 += 32) {
#pragma unroll
        for (int i = 0; i < 8; ++i) {
            const int idx = tid + i * 256;
            const int r = idx >> 5, c = idx & 31;
            As[r][c] = h1[(size_t)(m0 + r) * DFf + k0 + c];
        }
#pragma unroll
        for (int i = 0; i < 16; ++i) {
            const int idx = tid + i * 256;
            const int r = idx >> 7, c = idx & 127;
            Bs[r][c] = f2bf(B[(size_t)(k0 + r) * Hh + n0 + c]);
        }
        if (k0 + 32 < DFf)
            __builtin_prefetch(&B[(size_t)(k0 + 32) * Hh + n0 + (tid & 31) * 4], 0, 1);
        __syncthreads();
        v16bf af[2];
        af[0] = load_a_frag(&As[wm * 32][0], 40);
        af[1] = load_a_frag(&As[wm * 32 + 16][0], 40);
#pragma unroll
        for (int j = 0; j < 2; ++j) {
            v16bf bfrag = load_b_frag(&Bs[0][wn * 32 + j * 16], 136);
            acc[0][j] = wmma_bf16(af[0], bfrag, acc[0][j]);
            acc[1][j] = wmma_bf16(af[1], bfrag, acc[1][j]);
        }
        __syncthreads();
    }

    const int n = lane & 15, mh = (lane >> 4) * 8;
#pragma unroll
    for (int i = 0; i < 2; ++i)
#pragma unroll
        for (int j = 0; j < 2; ++j)
#pragma unroll
            for (int r = 0; r < 8; ++r) {
                const int row = m0 + wm * 32 + i * 16 + mh + r;
                const int col = n0 + wn * 32 + j * 16 + n;
                contrib[(size_t)row * Hh + col] = slotgate[row] * acc[i][j][r];
            }
}

// ---------------- 8. Final combine (pure write of d_out, fixed order) ----------------
__global__ void combine_kernel(const float* __restrict__ x1, const float* __restrict__ contrib,
                               const int* __restrict__ slot_of, float* __restrict__ out) {
    const int idx = blockIdx.x * 256 + threadIdx.x;   // over Mtok*Hh
    const int t = idx >> 10, c = idx & (Hh - 1);
    const int s0 = slot_of[t * 2], s1 = slot_of[t * 2 + 1];
    out[idx] = x1[idx] + contrib[(size_t)s0 * Hh + c] + contrib[(size_t)s1 * Hh + c];
}

// ---------------- host launcher ----------------
extern "C" void kernel_launch(void* const* d_in, const int* in_sizes, int n_in,
                              void* d_out, int out_size, void* d_ws, size_t ws_size,
                              hipStream_t stream) {
    (void)in_sizes; (void)n_in; (void)out_size; (void)ws_size;
    const float* x       = (const float*)d_in[0];
    const float* norm1_w = (const float*)d_in[1];
    const float* norm2_w = (const float*)d_in[2];
    const float* wq      = (const float*)d_in[3];
    const float* wk      = (const float*)d_in[4];
    const float* wv      = (const float*)d_in[5];
    const float* wo      = (const float*)d_in[6];
    const float* rot     = (const float*)d_in[7];
    const float* gate_w  = (const float*)d_in[8];
    const float* w1      = (const float*)d_in[9];
    const float* w2      = (const float*)d_in[10];
    const float* w3      = (const float*)d_in[11];
    float* out = (float*)d_out;

    // workspace carve-up (256B aligned)
    char* ws = (char*)d_ws;
    size_t p = 0;
    auto alloc = [&](size_t bytes) { char* r = ws + p; p = (p + bytes + 255) & ~(size_t)255; return r; };
    __bf16* n1bf     = (__bf16*)alloc((size_t)Mtok * Hh * 2);
    float*  qkv      = (float*) alloc((size_t)3 * Mtok * Hh * 4);
    float*  attn_out = (float*) alloc((size_t)Mtok * Hh * 4);
    float*  x1       = (float*) alloc((size_t)Mtok * Hh * 4);
    __bf16* n2bf     = (__bf16*)alloc((size_t)Mtok * Hh * 2);
    __bf16* h1       = (__bf16*)alloc((size_t)SLOTS * DFf * 2);
    float*  contrib  = (float*) alloc((size_t)SLOTS * Hh * 4);
    int*    topidx   = (int*)   alloc((size_t)Mtok * 2 * 4);
    float*  topg     = (float*) alloc((size_t)Mtok * 2 * 4);
    int*    slot_of  = (int*)   alloc((size_t)Mtok * 2 * 4);
    int*    tokens   = (int*)   alloc((size_t)SLOTS * 4);
    float*  slotgate = (float*) alloc((size_t)SLOTS * 4);
    int*    counts   = (int*)   alloc(Ee * 4);
    int*    cnt2     = (int*)   alloc(Ee * 4);
    int*    off      = (int*)   alloc((Ee + 1) * 4);

    float* qbuf = qkv;
    float* kbuf = qkv + (size_t)Mtok * Hh;
    float* vbuf = qkv + (size_t)2 * Mtok * Hh;

    // 1) n1 = rmsnorm(x) (bf16)
    rmsnorm_kernel<<<Mtok, 256, 0, stream>>>(x, norm1_w, n1bf);
    // 2) q,k,v = n1 @ {wq,wk,wv}
    gemm_kernel<__bf16, false><<<dim3(Hh / 128, Mtok / 64, 3), 256, 0, stream>>>(
        n1bf, wq, wk, wv, qkv, nullptr, Mtok, Hh, Hh);
    // 3) RoPE
    rope_kernel<<<(Bb * Ss * NHh * 32) / 256, 256, 0, stream>>>(qbuf, kbuf, rot);
    // 4) sliding-window flash attention
    attn_kernel<<<dim3(Ss / 16 / 8, NHh, Bb), 256, 0, stream>>>(qbuf, kbuf, vbuf, attn_out);
    // 5) x1 = x + attn_out @ wo
    gemm_kernel<float, true><<<dim3(Hh / 128, Mtok / 64, 1), 256, 0, stream>>>(
        attn_out, wo, wo, wo, x1, x, Mtok, Hh, Hh);
    // 6) n2 = rmsnorm(x1) (bf16)
    rmsnorm_kernel<<<Mtok, 256, 0, stream>>>(x1, norm2_w, n2bf);
    // 7) routing
    init_kernel<<<(SLOTS + 255) / 256, 256, 0, stream>>>(counts, cnt2, tokens);
    gate_kernel<<<Mtok / 8, 256, 0, stream>>>(n2bf, gate_w, topidx, topg, counts);
    scan_kernel<<<1, 32, 0, stream>>>(counts, off);
    scatter_kernel<<<(Mtok + 255) / 256, 256, 0, stream>>>(topidx, topg, off, cnt2,
                                                           tokens, slotgate, slot_of);
    // 8) expert FFN (top-2 only, expert-compacted)
    moe1_kernel<<<dim3(DFf / 128, SLOTS / 64), 256, 0, stream>>>(n2bf, tokens, off, w1, w3, h1);
    moe2_kernel<<<dim3(Hh / 128, SLOTS / 64), 256, 0, stream>>>(h1, off, w2, slotgate, contrib);
    // 9) out = x1 + gated expert contributions (deterministic order)
    combine_kernel<<<(Mtok * Hh) / 256, 256, 0, stream>>>(x1, contrib, slot_of, out);
}
